// GraphConv_33552284516567
// MI455X (gfx1250) — matrix operand order repaired and checked by reading, not compiled, not run on hardware
//
#include <hip/hip_runtime.h>

typedef __attribute__((ext_vector_type(16))) _Float16 v16h;
typedef __attribute__((ext_vector_type(8)))  float    v8f;
typedef __attribute__((ext_vector_type(4)))  unsigned int u32x4;
typedef __attribute__((ext_vector_type(2)))  unsigned int u32x2;

#define NB 8
#define NN 2048
#define FF 128
#define DD 128
#define MT 64     // rows per workgroup -> 256 WGs
#define KC 32     // K chunk (one wmma k-step)

// LDS layout (bytes)
#define OFF_W    0          // 128*128 f16 = 32768
#define OFF_H1   32768      // 64*128 f16  = 16384
#define OFF_BUF  49152      // 2 x (sA 64*32 f16 = 4096 + sX 128*32 f16 = 8192) = 24576
#define BUF_STRIDE 12288
#define OFF_H2   49152      // 64*128 f32 = 32768 (aliases staging buffers after GEMM1)
#define OFF_STAT 81920      // mu[64] rs[64] b[128] g[128] bt[128] = 512 floats
#define SMEM_BYTES (81920 + 512*4)

union FragU { v16h v; u32x4 u[2]; };
union H4U   { _Float16 h[4]; u32x2 u; };

// A-fragment (16x32 f16, MxK) from row-major LDS tile [m][k], stride in halves.
// ISA 7.12.2: lanes 0-15: M=lane, K={k0..k0+7, k0+16..k0+23};
//             lanes 16-31: M=lane-16, K={k0+8..k0+15, k0+24..k0+31}.
__device__ __forceinline__ v16h load_fragA(const _Float16* s, int m0, int strideH, int k0, int lane) {
    int l15 = lane & 15;
    int hi8 = (lane >> 4) << 3;
    const _Float16* base = s + (m0 + l15) * strideH + k0 + hi8;
    FragU f;
    f.u[0] = *(const u32x4*)(base);
    f.u[1] = *(const u32x4*)(base + 16);
    return f.v;
}

// B-fragment (32x16 f16, KxN) from column-major LDS tile [n][k], stride in halves.
// Lanes 0-15: N=lane, K={k0..k0+7, k0+16..k0+23}; lanes 16-31: N=lane-16, K={k0+8..k0+15, k0+24..k0+31}.
__device__ __forceinline__ v16h load_fragB(const _Float16* s, int n0, int strideH, int k0, int lane) {
    int l15  = lane & 15;
    int hi16 = (lane >> 4) << 3;
    const _Float16* base = s + (n0 + l15) * strideH + k0 + hi16;
    FragU f;
    f.u[0] = *(const u32x4*)(base);
    f.u[1] = *(const u32x4*)(base + 16);
    return f.v;
}

#define WMMA(a, b, c) __builtin_amdgcn_wmma_f32_16x16x32_f16(false, (a), false, (b), (short)0, (c), false, false)

__global__ void __launch_bounds__(256)
gcn_fused(const float* __restrict__ A, const float* __restrict__ X,
          const float* __restrict__ W, const float* __restrict__ bias,
          const float* __restrict__ gamma, const float* __restrict__ beta,
          float* __restrict__ out)
{
    extern __shared__ char smem[];
    _Float16* sW  = (_Float16*)(smem + OFF_W);
    _Float16* sH1 = (_Float16*)(smem + OFF_H1);
    float*    sH2 = (float*)   (smem + OFF_H2);
    float*    sMu = (float*)   (smem + OFF_STAT);
    float*    sRs = sMu + 64;
    float*    sB  = sMu + 128;
    float*    sG  = sMu + 256;
    float*    sBt = sMu + 384;

    const int tid   = threadIdx.x;
    const int wave  = tid >> 5;
    const int lane  = tid & 31;
    const int batch = blockIdx.x >> 5;           // 32 row-tiles per batch
    const int row0  = (blockIdx.x & 31) * MT;

    // ---- stage W (f32 -> f16) and the small vectors into LDS ----
    const float4* W4 = (const float4*)W;
    #pragma unroll
    for (int i = 0; i < 16; ++i) {
        int idx = i * 256 + tid;                 // 4096 float4
        float4 w = W4[idx];
        H4U p;
        p.h[0] = (_Float16)w.x; p.h[1] = (_Float16)w.y;
        p.h[2] = (_Float16)w.z; p.h[3] = (_Float16)w.w;
        *(u32x2*)(sW + idx * 4) = p.u;
    }
    if (tid < 128) { sB[tid] = bias[tid]; sG[tid] = gamma[tid]; sBt[tid] = beta[tid]; }

    // wave tile: 32 rows x 32 cols (2x4 wave grid over 64x128)
    const int m0 = (wave >> 2) * 32;
    const int n0 = (wave & 3) * 32;

    const float* Abase = A + ((size_t)batch * NN + row0) * NN;
    const float* Xbase = X + (size_t)batch * NN * FF;

    float4 ra[2], rx[4];

    auto loadRegs = [&](int kc) {
        #pragma unroll
        for (int i = 0; i < 2; ++i) {            // A chunk: 64x32 = 512 float4
            int idx = i * 256 + tid;
            int r = idx >> 3, kq = (idx & 7) * 4;
            ra[i] = *(const float4*)(Abase + (size_t)r * NN + kc + kq);
        }
        #pragma unroll
        for (int i = 0; i < 4; ++i) {            // X chunk: 32x128 = 1024 float4
            int idx = i * 256 + tid;
            int kr = idx >> 5, cq = (idx & 31) * 4;
            rx[i] = *(const float4*)(Xbase + (size_t)(kc + kr) * FF + cq);
        }
    };
    auto storeLds = [&](int bufIdx) {
        _Float16* sA = (_Float16*)(smem + OFF_BUF + bufIdx * BUF_STRIDE);
        _Float16* sX = sA + MT * KC;
        #pragma unroll
        for (int i = 0; i < 2; ++i) {
            int idx = i * 256 + tid;
            int r = idx >> 3, kq = (idx & 7) * 4;
            H4U p;
            p.h[0] = (_Float16)ra[i].x; p.h[1] = (_Float16)ra[i].y;
            p.h[2] = (_Float16)ra[i].z; p.h[3] = (_Float16)ra[i].w;
            *(u32x2*)(sA + r * KC + kq) = p.u;
        }
        #pragma unroll
        for (int i = 0; i < 4; ++i) {
            int idx = i * 256 + tid;
            int kr = idx >> 5, cq = (idx & 31) * 4;
            sX[(cq + 0) * KC + kr] = (_Float16)rx[i].x;
            sX[(cq + 1) * KC + kr] = (_Float16)rx[i].y;
            sX[(cq + 2) * KC + kr] = (_Float16)rx[i].z;
            sX[(cq + 3) * KC + kr] = (_Float16)rx[i].w;
        }
    };

    const v8f vzero = {0.f,0.f,0.f,0.f,0.f,0.f,0.f,0.f};
    v8f acc[2][2];
    #pragma unroll
    for (int i = 0; i < 2; ++i)
        #pragma unroll
        for (int j = 0; j < 2; ++j) acc[i][j] = vzero;

    // ====== GEMM1: H1 = A_hat[b, row0:row0+64, :] @ X[b]  (double-buffered) ======
    loadRegs(0);
    storeLds(0);
    __syncthreads();

    int buf = 0;
    for (int kc = 0; kc < NN; kc += KC, buf ^= 1) {
        const bool more = (kc + KC) < NN;
        if (more) loadRegs(kc + KC);             // issue next-chunk global loads

        const _Float16* sA = (const _Float16*)(smem + OFF_BUF + buf * BUF_STRIDE);
        const _Float16* sX = sA + MT * KC;
        v16h a0 = load_fragA(sA, m0,      KC, 0, lane);
        v16h a1 = load_fragA(sA, m0 + 16, KC, 0, lane);
        v16h b0 = load_fragB(sX, n0,      KC, 0, lane);
        v16h b1 = load_fragB(sX, n0 + 16, KC, 0, lane);

        acc[0][0] = WMMA(a0, b0, acc[0][0]);
        acc[0][1] = WMMA(a0, b1, acc[0][1]);
        acc[1][0] = WMMA(a1, b0, acc[1][0]);
        acc[1][1] = WMMA(a1, b1, acc[1][1]);

        if (more) storeLds(buf ^ 1);             // convert + stage into alternate buffer
        __syncthreads();
    }

    // ---- spill H1 (f16) to LDS per ISA C/D layout ----
    #pragma unroll
    for (int i = 0; i < 2; ++i)
        #pragma unroll
        for (int j = 0; j < 2; ++j)
            #pragma unroll
            for (int r = 0; r < 8; ++r) {
                int row = m0 + i * 16 + ((lane >> 4) << 3) + r;
                int col = n0 + j * 16 + (lane & 15);
                sH1[row * FF + col] = (_Float16)acc[i][j][r];
            }
    __syncthreads();

    // ====== GEMM2: H2 = H1 @ W^T  (W is [d][f] row-major == B col-major) ======
    #pragma unroll
    for (int i = 0; i < 2; ++i)
        #pragma unroll
        for (int j = 0; j < 2; ++j) acc[i][j] = vzero;

    #pragma unroll
    for (int k0 = 0; k0 < FF; k0 += KC) {
        v16h a0 = load_fragA(sH1, m0,      FF, k0, lane);
        v16h a1 = load_fragA(sH1, m0 + 16, FF, k0, lane);
        v16h b0 = load_fragB(sW, n0,      FF, k0, lane);
        v16h b1 = load_fragB(sW, n0 + 16, FF, k0, lane);

        acc[0][0] = WMMA(a0, b0, acc[0][0]);
        acc[0][1] = WMMA(a0, b1, acc[0][1]);
        acc[1][0] = WMMA(a1, b0, acc[1][0]);
        acc[1][1] = WMMA(a1, b1, acc[1][1]);
    }

    // ---- H2 + bias into LDS f32 (aliases staging buffers; GEMM2 read only sH1/sW) ----
    #pragma unroll
    for (int i = 0; i < 2; ++i)
        #pragma unroll
        for (int j = 0; j < 2; ++j)
            #pragma unroll
            for (int r = 0; r < 8; ++r) {
                int row = m0 + i * 16 + ((lane >> 4) << 3) + r;
                int col = n0 + j * 16 + (lane & 15);
                sH2[row * DD + col] = acc[i][j][r] + sB[col];
            }
    __syncthreads();

    // ====== LayerNorm stats (one thread per row) ======
    if (tid < MT) {
        const float* rowp = sH2 + tid * DD;
        float s = 0.f;
        #pragma unroll
        for (int c = 0; c < DD; c += 4) {
            float4 v = *(const float4*)(rowp + c);
            s += v.x + v.y + v.z + v.w;
        }
        float mu = s * (1.f / 128.f);
        float vs = 0.f;
        #pragma unroll
        for (int c = 0; c < DD; c += 4) {
            float4 v = *(const float4*)(rowp + c);
            float dx = v.x - mu, dy = v.y - mu, dz = v.z - mu, dw = v.w - mu;
            vs += dx * dx + dy * dy + dz * dz + dw * dw;
        }
        sMu[tid] = mu;
        sRs[tid] = rsqrtf(vs * (1.f / 128.f) + 1e-5f);
    }
    __syncthreads();

    // ====== normalize + gamma/beta + ReLU, fully coalesced store ======
    float* outBase = out + ((size_t)batch * NN + row0) * DD;
    #pragma unroll
    for (int i = 0; i < 32; ++i) {
        int idx = i * 256 + tid;                 // 0..8191
        int r = idx >> 7;
        int d = idx & 127;
        float v = (sH2[idx] - sMu[r]) * sRs[r] * sG[d] + sBt[d];
        outBase[idx] = fmaxf(v, 0.f);
    }
}

extern "C" void kernel_launch(void* const* d_in, const int* in_sizes, int n_in,
                              void* d_out, int out_size, void* d_ws, size_t ws_size,
                              hipStream_t stream) {
    const float* A   = (const float*)d_in[0];
    const float* X   = (const float*)d_in[1];
    const float* W   = (const float*)d_in[2];
    const float* b   = (const float*)d_in[3];
    const float* gam = (const float*)d_in[4];
    const float* bet = (const float*)d_in[5];
    float* out       = (float*)d_out;

    hipFuncSetAttribute((const void*)gcn_fused,
                        hipFuncAttributeMaxDynamicSharedMemorySize, (int)SMEM_BYTES);
    dim3 grid(NB * (NN / MT));   // 256 workgroups
    gcn_fused<<<grid, 256, SMEM_BYTES, stream>>>(A, X, W, b, gam, bet, out);
}
